// Model_532575944714
// MI455X (gfx1250) — matrix operand order repaired and checked by reading, not compiled, not run on hardware
//
#include <hip/hip_runtime.h>
#include <math.h>

// ---------------------------------------------------------------------------
// Types / helpers
// ---------------------------------------------------------------------------
typedef _Float16 v16h __attribute__((ext_vector_type(16)));
typedef _Float16 v8h  __attribute__((ext_vector_type(8)));
typedef _Float16 v4h  __attribute__((ext_vector_type(4)));
typedef float    v8f  __attribute__((ext_vector_type(8)));

#define WMMA16(A,B,C) __builtin_amdgcn_wmma_f32_16x16x32_f16(false,(A),false,(B),(short)0,(C),false,false)

__device__ __forceinline__ v8f zero_v8f(){ v8f z;
#pragma unroll
  for (int i=0;i<8;++i) z[i]=0.f; return z; }

__device__ __forceinline__ float ffreq(int m, int N){
  return (m < (N>>1) ? (float)m : (float)(m - N)) / (float)N;
}

// ---------------------------------------------------------------------------
// Constant generation kernels (deterministic; regenerated every launch)
// ---------------------------------------------------------------------------
__global__ void k_gen_dft(_Float16* fr, _Float16* fi, int N, float sign, float scale){
  int i = blockIdx.x*256 + threadIdx.x;
  if (i >= N*N) return;
  int m = i / N, n = i % N;
  int r = (m*n) % N;                       // exact integer phase reduction
  float th = sign * 6.2831853071795864f * (float)r / (float)N;
  fr[i] = (_Float16)(cosf(th)*scale);
  fi[i] = (_Float16)(sinf(th)*scale);
}

// Morlet band-pass filters in Fourier domain (real-valued), L orientations.
__global__ void k_gen_psi(_Float16* out, int N, int L, float xi, float sigma, float slant){
  int i = blockIdx.x*256 + threadIdx.x;
  int NN = N*N;
  if (i >= L*NN) return;
  int l = i / NN, rem = i - l*NN;
  int m = rem / N, n = rem - m*N;
  float theta = (float)l * 3.14159265358979f / (float)L;
  float wx = 6.2831853071795864f * ffreq(m, N);
  float wy = 6.2831853071795864f * ffreq(n, N);
  float ct = cosf(theta), st = sinf(theta);
  float u =  ct*wx + st*wy;
  float v = -st*wx + ct*wy;
  float vs = v / slant;
  float s2 = 0.5f * sigma * sigma;
  float gab   = expf(-s2*((u - xi)*(u - xi) + vs*vs));
  float gauss = expf(-s2*(u*u + vs*vs));
  float kappa = expf(-s2*xi*xi);
  out[i] = (_Float16)(gab - kappa*gauss);
}

// Low-pass Gaussian, optionally pre-folded (periodized) from Nout*fold grid.
__global__ void k_gen_phi(_Float16* out, int Nout, int fold, float sigma){
  int i = blockIdx.x*256 + threadIdx.x;
  if (i >= Nout*Nout) return;
  int a = i / Nout, b = i - a*Nout;
  int Nfull = Nout*fold;
  float s2 = 0.5f * sigma * sigma;
  float acc = 0.f;
  for (int p = 0; p < fold; ++p)
    for (int q = 0; q < fold; ++q){
      float wx = 6.2831853071795864f * ffreq(p*Nout + a, Nfull);
      float wy = 6.2831853071795864f * ffreq(q*Nout + b, Nfull);
      acc += expf(-s2*(wx*wx + wy*wy));
    }
  out[i] = (_Float16)(acc / (float)(fold*fold));
}

__global__ void k_f32_to_f16(const float* in, _Float16* out, long long n){
  long long i = (long long)blockIdx.x*256 + threadIdx.x;
  if (i < n) out[i] = (_Float16)in[i];
}

// ---------------------------------------------------------------------------
// Pointwise spectral multiply by a real filter + Fourier-domain fold (mean of
// kxk blocks).  Expands batch by L (one output image per orientation).
// ---------------------------------------------------------------------------
__global__ void k_mul_fold(const _Float16* __restrict__ inR, const _Float16* __restrict__ inI,
                           int Nin, const _Float16* __restrict__ filt, int L, int fold,
                           _Float16* __restrict__ outR, _Float16* __restrict__ outI,
                           long long total, int batchIn){
  long long idx = (long long)blockIdx.x*256 + threadIdx.x;
  if (idx >= total) return;
  int No = Nin / fold;
  int x = (int)(idx % No);
  long long t1 = idx / No;
  int y = (int)(t1 % No);
  long long t2 = t1 / No;
  int l = (int)(t2 % L);
  int a = (int)(t2 / L);
  float re = 0.f, im = 0.f;
  const _Float16* ir = inR + (long long)a*Nin*Nin;
  const _Float16* ii = inI + (long long)a*Nin*Nin;
  const _Float16* fl = filt + (long long)l*Nin*Nin;
  for (int p = 0; p < fold; ++p)
    for (int q = 0; q < fold; ++q){
      int off = (p*No + y)*Nin + (q*No + x);
      float f = (float)fl[off];
      re += (float)ir[off] * f;
      im += (float)ii[off] * f;
    }
  float inv = 1.f / (float)(fold*fold);
  long long o = ((long long)a*L + l)*No*No + (long long)y*No + x;
  outR[o] = (_Float16)(re*inv);
  outI[o] = (_Float16)(im*inv);
}

__global__ void k_modulus(const _Float16* re, const _Float16* im, _Float16* out, long long n){
  long long i = (long long)blockIdx.x*256 + threadIdx.x;
  if (i >= n) return;
  float r = (float)re[i], m = (float)im[i];
  out[i] = (_Float16)sqrtf(r*r + m*m);
}

// ---------------------------------------------------------------------------
// Batched complex GEMM, C = A x B, planar complex f16, f32 accumulation via
// v_wmma_f32_16x16x32_f16.  A is MxK (lda), B is KxN (ldb).  Template params
// AI/BI say whether the imaginary plane of A/B exists (compile-time: no
// branches around WMMA in the K loop).  stride==0 => operand shared across
// the batch (the DFT matrix).  Output: complex f16 (CrH/CiH) or real f32
// (CrF) scattered into the scattering tensor via (outLp, outChanBase, outKTot).
// Block = 128 threads = 4 waves; each wave owns one 16x16 C tile (block tile
// 16M x 64N), K consumed 32 at a time through LDS (B transposed to [n][k]).
// ---------------------------------------------------------------------------
template<bool AI, bool BI>
__launch_bounds__(128)
__global__ void k_cgemm(const _Float16* __restrict__ Ar, const _Float16* __restrict__ Ai, long long sA,
                        const _Float16* __restrict__ Br, const _Float16* __restrict__ Bi, long long sB,
                        _Float16* __restrict__ CrH, _Float16* __restrict__ CiH,
                        float* __restrict__ CrF, long long sC,
                        int M, int N, int K, int lda, int ldb, int ldc,
                        int outLp, int outChanBase, int outKTot){
  __shared__ __align__(32) _Float16 shA[2*16*32];   // [plane][m][k]
  __shared__ __align__(32) _Float16 shB[2*64*32];   // [plane][n][k]  (transposed)
  _Float16* sAr = shA;            _Float16* sAi = shA + 16*32;
  _Float16* sBr = shB;            _Float16* sBi = shB + 64*32;

  const int t    = threadIdx.x;
  const int lane = t & 31;
  const int wv   = t >> 5;
  const int bat  = blockIdx.z;
  const int mBase = blockIdx.y << 4;
  const int nBlk  = blockIdx.x << 6;

  const _Float16* gAr = Ar + (long long)bat * sA;
  const _Float16* gAi = AI ? Ai + (long long)bat * sA : nullptr;
  const _Float16* gBr = Br + (long long)bat * sB;
  const _Float16* gBi = BI ? Bi + (long long)bat * sB : nullptr;

  v8f accP = zero_v8f();   // Ar*Br
  v8f accM = zero_v8f();   // Ai*Bi  (subtracted at the end)
  v8f accI = zero_v8f();   // Ar*Bi + Ai*Br

  const int mfr = lane & 15;
  const int hi  = lane >> 4;

  for (int k0 = 0; k0 < K; k0 += 32){
    // ---- stage A tile (16 x 32), 4 contiguous halves per thread per plane
    {
      const int idx = t << 2;
      const int row = idx >> 5, col = idx & 31;
      const int gm = mBase + row;
      if (gm < M && (k0 + col + 3) < K){
        *(v4h*)&sAr[idx] = *(const v4h*)&gAr[(long long)gm*lda + k0 + col];
        if (AI) *(v4h*)&sAi[idx] = *(const v4h*)&gAi[(long long)gm*lda + k0 + col];
      } else {
#pragma unroll
        for (int j = 0; j < 4; ++j){
          int kk = k0 + col + j;
          bool ok = (gm < M) && (kk < K);
          sAr[idx+j] = ok ? gAr[(long long)gm*lda + kk] : (_Float16)0.f;
          if (AI) sAi[idx+j] = ok ? gAi[(long long)gm*lda + kk] : (_Float16)0.f;
        }
      }
    }
    // ---- stage B tile (32 x 64) transposed into [n][k]
    if ((k0 + 32 <= K) && (nBlk + 64 <= N)){
      // interior fast path: 8 halves per global_load_b128, scatter to LDS
#pragma unroll
      for (int p = 0; p < 2; ++p){
        int i  = (p*128 + t) << 3;       // half index, multiple of 8
        int kk = i >> 6, nn = i & 63;
        v8h d = *(const v8h*)&gBr[(long long)(k0+kk)*ldb + nBlk + nn];
#pragma unroll
        for (int j = 0; j < 8; ++j) sBr[(nn+j)*32 + kk] = d[j];
        if (BI){
          v8h e = *(const v8h*)&gBi[(long long)(k0+kk)*ldb + nBlk + nn];
#pragma unroll
          for (int j = 0; j < 8; ++j) sBi[(nn+j)*32 + kk] = e[j];
        }
      }
    } else {
#pragma unroll
      for (int p = 0; p < 16; ++p){
        int i  = p*128 + t;              // 0..2047
        int kk = i >> 6, nn = i & 63;
        int gk = k0 + kk, gn = nBlk + nn;
        bool ok = (gk < K) && (gn < N);
        sBr[nn*32 + kk] = ok ? gBr[(long long)gk*ldb + gn] : (_Float16)0.f;
        if (BI) sBi[nn*32 + kk] = ok ? gBi[(long long)gk*ldb + gn] : (_Float16)0.f;
      }
    }
    __syncthreads();

    // ---- gather fragments; k order 16*hi + s, same for A and B
    const int aoff = mfr*32 + (hi << 4);           // A: lane holds row M, contiguous k
    v8h alo = *(const v8h*)&sAr[aoff];
    v8h ahi = *(const v8h*)&sAr[aoff + 8];
    v16h fAr = __builtin_shufflevector(alo, ahi, 0,1,2,3,4,5,6,7,8,9,10,11,12,13,14,15);

    const int boff = ((wv << 4) + mfr)*32 + (hi << 4);  // B: lane holds col N, contiguous k
    v16h fBr = *(const v16h*)&sBr[boff];

    accP = WMMA16(fAr, fBr, accP);
    if (AI){
      v8h ilo = *(const v8h*)&sAi[aoff];
      v8h ihi = *(const v8h*)&sAi[aoff + 8];
      v16h fAi = __builtin_shufflevector(ilo, ihi, 0,1,2,3,4,5,6,7,8,9,10,11,12,13,14,15);
      accI = WMMA16(fAi, fBr, accI);
      if (BI){
        v16h fBi = *(const v16h*)&sBi[boff];
        accM = WMMA16(fAi, fBi, accM);
        accI = WMMA16(fAr, fBi, accI);
      }
    } else if (BI){
      v16h fBi = *(const v16h*)&sBi[boff];
      accI = WMMA16(fAr, fBi, accI);
    }
    __syncthreads();
  }

  // ---- epilogue: C layout: VGPR v -> M = v + 8*hi ; N = lane&15 per wave tile
  v8f accR;
  if (AI && BI) accR = accP - accM; else accR = accP;
  const int nOut = nBlk + (wv << 4) + mfr;

  if (CrF){
    const int bc = bat / outLp;
    const int l  = bat - bc*outLp;
    float* cp = CrF + ((long long)bc*outKTot + outChanBase + l) * sC;
#pragma unroll
    for (int v = 0; v < 8; ++v){
      int mm = mBase + v + (hi << 3);
      if (mm < M && nOut < N) cp[(long long)mm*ldc + nOut] = accR[v];
    }
  } else {
    _Float16* pr = CrH + (long long)bat*sC;
    _Float16* pi = CiH ? CiH + (long long)bat*sC : nullptr;
#pragma unroll
    for (int v = 0; v < 8; ++v){
      int mm = mBase + v + (hi << 3);
      if (mm < M && nOut < N){
        pr[(long long)mm*ldc + nOut] = (_Float16)accR[v];
        if (pi) pi[(long long)mm*ldc + nOut] = (_Float16)accI[v];
      }
    }
  }
}

// ---------------------------------------------------------------------------
// Global max pool over 56x56 per (image, channel); reorders (bc,k) -> b*243+c*81+k
// ---------------------------------------------------------------------------
__global__ void k_maxpool(const float* __restrict__ S, float* __restrict__ G){
  __shared__ float red[256];
  int blk = blockIdx.x;                    // bc*81 + k, bc in [0,48)
  const float* p = S + (long long)blk*3136;
  float m = -3.4e38f;
  for (int i = threadIdx.x; i < 3136; i += 256) m = fmaxf(m, p[i]);
  red[threadIdx.x] = m;
  __syncthreads();
  for (int s = 128; s > 0; s >>= 1){
    if (threadIdx.x < s) red[threadIdx.x] = fmaxf(red[threadIdx.x], red[threadIdx.x+s]);
    __syncthreads();
  }
  if (threadIdx.x == 0){
    int bc = blk / 81, k = blk - bc*81;
    int b = bc / 3,   c = bc - b*3;
    G[b*243 + c*81 + k] = red[0];
  }
}

// out[b,o] = sum_c g[b,c] * W[o,c] + bias[o]     (tiny: 7.8 MFLOP)
__global__ void k_linear(const float* __restrict__ G, const float* __restrict__ W,
                         const float* __restrict__ bias, float* __restrict__ out){
  int i = blockIdx.x*256 + threadIdx.x;
  if (i >= 16*1000) return;
  int b = i / 1000, o = i - b*1000;
  float acc = bias[o];
  const float* g = G + b*243;
  const float* w = W + o*243;
  for (int c = 0; c < 243; ++c) acc += g[c]*w[c];
  out[i] = acc;
}

// ---------------------------------------------------------------------------
// Host driver
// ---------------------------------------------------------------------------
extern "C" void kernel_launch(void* const* d_in, const int* in_sizes, int n_in,
                              void* d_out, int out_size, void* d_ws, size_t ws_size,
                              hipStream_t stream){
  (void)in_sizes; (void)n_in; (void)out_size; (void)ws_size;
  const float* x  = (const float*)d_in[0];     // (16,3,224,224)
  const float* Wl = (const float*)d_in[1];     // (1000,243)
  const float* bl = (const float*)d_in[2];     // (1000,)
  float* out = (float*)d_out;                  // (16,1000)

  const long long P224 = 224*224, P112 = 112*112, P56 = 56*56;
  const int BC = 48;

  // bump allocator over d_ws
  char* ws = (char*)d_ws;
  size_t off = 0;
  auto alloc = [&](size_t bytes)->char*{
    off = (off + 255) & ~(size_t)255;
    char* p = ws + off; off += bytes; return p;
  };
  auto a16 = [&](long long n)->_Float16*{ return (_Float16*)alloc((size_t)n*2); };
  auto af  = [&](long long n)->float*   { return (float*)   alloc((size_t)n*4); };

  _Float16 *F224r=a16(P224), *F224i=a16(P224), *B224r=a16(P224), *B224i=a16(P224);
  _Float16 *F112r=a16(P112), *F112i=a16(P112), *B112r=a16(P112), *B112i=a16(P112);
  _Float16 *F56r =a16(P56),  *F56i =a16(P56),  *B56r =a16(P56),  *B56i =a16(P56);
  _Float16 *PSI0 =a16(8*P224), *PSI1 =a16(8*P224), *PHI0 =a16(P224), *PHI1 =a16(P112);
  _Float16 *XR   =a16((long long)BC*P224);
  _Float16 *XHr  =a16((long long)BC*P224), *XHi=a16((long long)BC*P224);
  _Float16 *T224r=a16(384LL*P224), *T224i=a16(384LL*P224);
  _Float16 *U224r=a16(384LL*P224), *U224i=a16(384LL*P224);
  _Float16 *M224 =a16(384LL*P224);
  _Float16 *V112r=a16(3072LL*P112), *V112i=a16(3072LL*P112);
  _Float16 *W112r=a16(3072LL*P112), *W112i=a16(3072LL*P112);
  _Float16 *M112 =a16(3072LL*P112);
  _Float16 *P56r =a16(3072LL*P56),  *P56i =a16(3072LL*P56);
  _Float16 *Q56r =a16(3072LL*P56),  *Q56i =a16(3072LL*P56);
  float    *S    =af((long long)BC*81*P56);
  float    *G    =af(16*243);

  auto gemm = [&](const _Float16* Ar,const _Float16* Ai,long long sA,
                  const _Float16* Br,const _Float16* Bi,long long sB,
                  _Float16* Cr,_Float16* Ci,float* Cf,long long sC,
                  int M,int N,int K,int lda,int ldb,int ldc,int batch,
                  int Lp,int cb,int kt){
    dim3 g((N+63)/64, (M+15)/16, batch);
    if (Ai && Bi)
      k_cgemm<true,true ><<<g,dim3(128),0,stream>>>(Ar,Ai,sA,Br,Bi,sB,Cr,Ci,Cf,sC,M,N,K,lda,ldb,ldc,Lp,cb,kt);
    else if (Ai)
      k_cgemm<true,false><<<g,dim3(128),0,stream>>>(Ar,Ai,sA,Br,Bi,sB,Cr,Ci,Cf,sC,M,N,K,lda,ldb,ldc,Lp,cb,kt);
    else if (Bi)
      k_cgemm<false,true><<<g,dim3(128),0,stream>>>(Ar,Ai,sA,Br,Bi,sB,Cr,Ci,Cf,sC,M,N,K,lda,ldb,ldc,Lp,cb,kt);
    else
      k_cgemm<false,false><<<g,dim3(128),0,stream>>>(Ar,Ai,sA,Br,Bi,sB,Cr,Ci,Cf,sC,M,N,K,lda,ldb,ldc,Lp,cb,kt);
  };
  auto mulfold = [&](const _Float16* ir,const _Float16* ii,int Nin,
                     const _Float16* f,int Lf,int fold,
                     _Float16* orr,_Float16* oi,int batchIn){
    int No = Nin/fold;
    long long tot = (long long)batchIn*Lf*No*No;
    k_mul_fold<<<(int)((tot+255)/256), 256, 0, stream>>>(ir,ii,Nin,f,Lf,fold,orr,oi,tot,batchIn);
  };

  // ---- constants -------------------------------------------------------
  const float XI0 = 2.35619449019234f;     // 3*pi/4
  const float SIG0 = 0.8f, SLANT = 0.5f, SIGPHI = 1.6f;
  k_gen_dft<<<(int)((P224+255)/256),256,0,stream>>>(F224r,F224i,224,-1.f,1.f);
  k_gen_dft<<<(int)((P224+255)/256),256,0,stream>>>(B224r,B224i,224,+1.f,1.f/224.f);
  k_gen_dft<<<(int)((P112+255)/256),256,0,stream>>>(F112r,F112i,112,-1.f,1.f);
  k_gen_dft<<<(int)((P112+255)/256),256,0,stream>>>(B112r,B112i,112,+1.f,1.f/112.f);
  k_gen_dft<<<(int)((P56 +255)/256),256,0,stream>>>(F56r, F56i, 56,-1.f,1.f);
  k_gen_dft<<<(int)((P56 +255)/256),256,0,stream>>>(B56r, B56i, 56,+1.f,1.f/56.f);
  k_gen_psi<<<(int)((8*P224+255)/256),256,0,stream>>>(PSI0,224,8, XI0,      SIG0,      SLANT);
  k_gen_psi<<<(int)((8*P224+255)/256),256,0,stream>>>(PSI1,224,8, XI0*0.5f, SIG0*2.f,  SLANT);
  k_gen_phi<<<(int)((P224+255)/256),256,0,stream>>>(PHI0,224,1,SIGPHI);
  k_gen_phi<<<(int)((P112+255)/256),256,0,stream>>>(PHI1,112,2,SIGPHI);

  // ---- xh = fft2(x) ----------------------------------------------------
  k_f32_to_f16<<<(int)((BC*P224+255)/256),256,0,stream>>>(x, XR, (long long)BC*P224);
  gemm(F224r,F224i,0,  XR,nullptr,P224,  T224r,T224i,nullptr,P224, 224,224,224,224,224,224, BC, 1,0,0);
  gemm(T224r,T224i,P224, F224r,F224i,0,  XHr,XHi,nullptr,P224,    224,224,224,224,224,224, BC, 1,0,0);

  // ---- S0 : (xh*phi0) fold4 -> ifft2@56 -> channel 0 -------------------
  mulfold(XHr,XHi,224, PHI0,1,4, P56r,P56i, BC);
  gemm(B56r,B56i,0, P56r,P56i,P56, Q56r,Q56i,nullptr,P56, 56,56,56,56,56,56, BC, 1,0,0);
  gemm(Q56r,Q56i,P56, B56r,B56i,0, nullptr,nullptr,S,P56, 56,56,56,56,56,56, BC, 1,0,81);

  // ---- First order j1=0 : U = |ifft2(xh*psi0)| , U1H = fft2(U) ---------
  mulfold(XHr,XHi,224, PSI0,8,1, U224r,U224i, BC);                       // 384@224
  gemm(B224r,B224i,0, U224r,U224i,P224, T224r,T224i,nullptr,P224, 224,224,224,224,224,224, 384, 1,0,0);
  gemm(T224r,T224i,P224, B224r,B224i,0, U224r,U224i,nullptr,P224, 224,224,224,224,224,224, 384, 1,0,0);
  k_modulus<<<(int)((384*P224+255)/256),256,0,stream>>>(U224r,U224i,M224, 384LL*P224);
  gemm(F224r,F224i,0, M224,nullptr,P224, T224r,T224i,nullptr,P224, 224,224,224,224,224,224, 384, 1,0,0);
  gemm(T224r,T224i,P224, F224r,F224i,0, U224r,U224i,nullptr,P224, 224,224,224,224,224,224, 384, 1,0,0);

  // S1 (j1=0): channels 1..8
  mulfold(U224r,U224i,224, PHI0,1,4, P56r,P56i, 384);
  gemm(B56r,B56i,0, P56r,P56i,P56, Q56r,Q56i,nullptr,P56, 56,56,56,56,56,56, 384, 1,0,0);
  gemm(Q56r,Q56i,P56, B56r,B56i,0, nullptr,nullptr,S,P56, 56,56,56,56,56,56, 384, 8,1,81);

  // Second order (j1=0, j2=1): channels 17..80
  mulfold(U224r,U224i,224, PSI1,8,2, V112r,V112i, 384);                  // 3072@112
  gemm(B112r,B112i,0, V112r,V112i,P112, W112r,W112i,nullptr,P112, 112,112,112,112,112,112, 3072, 1,0,0);
  gemm(W112r,W112i,P112, B112r,B112i,0, V112r,V112i,nullptr,P112, 112,112,112,112,112,112, 3072, 1,0,0);
  k_modulus<<<(int)((3072*P112+255)/256),256,0,stream>>>(V112r,V112i,M112, 3072LL*P112);
  gemm(F112r,F112i,0, M112,nullptr,P112, W112r,W112i,nullptr,P112, 112,112,112,112,112,112, 3072, 1,0,0);
  gemm(W112r,W112i,P112, F112r,F112i,0, V112r,V112i,nullptr,P112, 112,112,112,112,112,112, 3072, 1,0,0);
  mulfold(V112r,V112i,112, PHI1,1,2, P56r,P56i, 3072);
  gemm(B56r,B56i,0, P56r,P56i,P56, Q56r,Q56i,nullptr,P56, 56,56,56,56,56,56, 3072, 1,0,0);
  gemm(Q56r,Q56i,P56, B56r,B56i,0, nullptr,nullptr,S,P56, 56,56,56,56,56,56, 3072, 64,17,81);

  // ---- First order j1=1 : channels 9..16 -------------------------------
  mulfold(XHr,XHi,224, PSI1,8,2, V112r,V112i, BC);                       // 384@112
  gemm(B112r,B112i,0, V112r,V112i,P112, W112r,W112i,nullptr,P112, 112,112,112,112,112,112, 384, 1,0,0);
  gemm(W112r,W112i,P112, B112r,B112i,0, V112r,V112i,nullptr,P112, 112,112,112,112,112,112, 384, 1,0,0);
  k_modulus<<<(int)((384*P112+255)/256),256,0,stream>>>(V112r,V112i,M112, 384LL*P112);
  gemm(F112r,F112i,0, M112,nullptr,P112, W112r,W112i,nullptr,P112, 112,112,112,112,112,112, 384, 1,0,0);
  gemm(W112r,W112i,P112, F112r,F112i,0, V112r,V112i,nullptr,P112, 112,112,112,112,112,112, 384, 1,0,0);
  mulfold(V112r,V112i,112, PHI1,1,2, P56r,P56i, 384);
  gemm(B56r,B56i,0, P56r,P56i,P56, Q56r,Q56i,nullptr,P56, 56,56,56,56,56,56, 384, 1,0,0);
  gemm(Q56r,Q56i,P56, B56r,B56i,0, nullptr,nullptr,S,P56, 56,56,56,56,56,56, 384, 8,9,81);

  // ---- max pool + classifier ------------------------------------------
  k_maxpool<<<BC*81, 256, 0, stream>>>(S, G);
  k_linear<<<(16*1000+255)/256, 256, 0, stream>>>(G, Wl, bl, out);
}
